// CCIM_81982335746478
// MI455X (gfx1250) — matrix operand order repaired and checked by reading, not compiled, not run on hardware
//
#include <hip/hip_runtime.h>

// ---------------------------------------------------------------------------
// CDNA5 / gfx1250 fused CCIM (causal-intervention attention) pipeline.
// GEMMs via v_wmma_f32_16x16x32_bf16 (wave32), f32 accumulate.
// All operands pre-staged as bf16 (A row-major, B pre-transposed) so every
// WMMA operand load is a contiguous b128; Q block async-staged to LDS via
// global_load_async_to_lds_b128 + s_wait_asynccnt. GEMM inner loops use
// copy-free ping-pong prefetch (two half-steps per iteration) so loads
// overlap the XDL pipe without register rotation/spills.
// B=1024, K=256, D=1024, P=768, scale = 1/sqrt(1024) = 1/32.
// ---------------------------------------------------------------------------

typedef __attribute__((ext_vector_type(16))) __bf16 v16bf;
typedef __attribute__((ext_vector_type(8)))  __bf16 v8bf;
typedef __attribute__((ext_vector_type(8)))  float  v8f;

#define DIM_B 1024
#define DIM_K 256
#define DIM_D 1024
#define DIM_P 768

__device__ __forceinline__ v8f wmma_bf16(v16bf a, v16bf b, v8f c) {
  return __builtin_amdgcn_wmma_f32_16x16x32_bf16(
      false, a, false, b, (short)0, c, false, false);
}

__device__ __forceinline__ v16bf cat8(v8bf lo, v8bf hi) {
  return __builtin_shufflevector(lo, hi, 0, 1, 2, 3, 4, 5, 6, 7,
                                 8, 9, 10, 11, 12, 13, 14, 15);
}

// A operand: row 'row' of row-major bf16 matrix. Lanes 0-15: K {0..7,16..23};
// lanes 16-31: K {8..15,24..31}  -> two 16B contiguous runs per lane.
__device__ __forceinline__ v16bf loadA_bf16(const __bf16* X, int ld, int row,
                                            int k0, int kh) {
  const __bf16* base = X + row * ld + k0 + 8 * kh;
  return cat8(*(const v8bf*)base, *(const v8bf*)(base + 16));
}

// B operand (NT): B rows are rows of a row-major (pre-transposed) matrix;
// N = lane&15. Lanes 0-15: K {0..7,16..23}; lanes 16-31: K {8..15,24..31}.
__device__ __forceinline__ v16bf loadB_rows_bf16(const __bf16* X, int ld,
                                                 int row, int k0, int kh) {
  const __bf16* base = X + row * ld + k0 + 8 * kh;
  return cat8(*(const v8bf*)base, *(const v8bf*)(base + 16));
}

// ---------------------------------------------------------------------------
// Prep kernels (one-shot, coalesced reads, L2-resident outputs):
//   Jb[b,d] = bf16(jf[b,d])                          1024x1024
//   Wt[p,d] = bf16(W[d,p])                           768x1024 (per weight)
//   Cb[k,d] = bf16(conf[k,d]); Vt[d,k] = Cb*prior    1024x256
// ---------------------------------------------------------------------------
__global__ void ccim_cvt(const float* __restrict__ X, __bf16* __restrict__ Y) {
  int i = blockIdx.x * 256 + threadIdx.x;
  Y[i] = (__bf16)X[i];
}

__global__ void ccim_wprep(const float* __restrict__ W,
                           __bf16* __restrict__ Wt) {
  int i = blockIdx.x * 256 + threadIdx.x;          // 0 .. 786431
  int d = i / DIM_P, p = i - d * DIM_P;            // coalesced read of W
  Wt[p * DIM_D + d] = (__bf16)W[i];
}

__global__ void ccim_vprep(const float* __restrict__ conf,
                           const float* __restrict__ prior,
                           __bf16* __restrict__ Vt,
                           __bf16* __restrict__ Cb) {
  int i = blockIdx.x * 256 + threadIdx.x;          // 0 .. 262143
  int k = i >> 10, d = i & (DIM_D - 1);            // coalesced read of conf
  float c = conf[i];
  Cb[i] = (__bf16)c;                               // A operand for K-proj
  Vt[d * DIM_K + k] = (__bf16)(c * prior[k]);      // B operand for attn@V
}

// ---------------------------------------------------------------------------
// Projection: out[M,768] = bf16( Ab[M,1024] @ W[1024,768] ), W given as Wt.
// One wave -> 16x64 tile; ping-pong over two 32-wide half-steps per iter.
// ---------------------------------------------------------------------------
__global__ void ccim_proj(const __bf16* __restrict__ Ab,
                          const __bf16* __restrict__ Wt,
                          __bf16* __restrict__ out, int M) {
  const int wid  = blockIdx.x * (blockDim.x >> 5) + (threadIdx.x >> 5);
  const int lane = threadIdx.x & 31;
  const int mt = wid / 12, ns = wid % 12;          // 12 = 768/64 col supertiles
  if (mt * 16 >= M) return;
  const int row0 = mt * 16, col0 = ns * 64;
  const int n = lane & 15, kh = lane >> 4;

  v8f acc[4] = {};
  for (int k0 = 0; k0 < DIM_D; k0 += 64) {         // 16 iters, no remainder
    // half-step 0 loads
    v16bf a0 = loadA_bf16(Ab, DIM_D, row0 + n, k0, kh);
    v16bf b0[4];
#pragma unroll
    for (int t = 0; t < 4; ++t)
      b0[t] = loadB_rows_bf16(Wt, DIM_D, col0 + 16 * t + n, k0, kh);
    // half-step 1 loads issued before half-step 0 compute
    v16bf a1 = loadA_bf16(Ab, DIM_D, row0 + n, k0 + 32, kh);
    v16bf b1[4];
#pragma unroll
    for (int t = 0; t < 4; ++t)
      b1[t] = loadB_rows_bf16(Wt, DIM_D, col0 + 16 * t + n, k0 + 32, kh);
#pragma unroll
    for (int t = 0; t < 4; ++t) acc[t] = wmma_bf16(a0, b0[t], acc[t]);
#pragma unroll
    for (int t = 0; t < 4; ++t) acc[t] = wmma_bf16(a1, b1[t], acc[t]);
  }

  const int mo = kh * 8;
#pragma unroll
  for (int t = 0; t < 4; ++t)
#pragma unroll
    for (int r = 0; r < 8; ++r)
      out[(row0 + mo + r) * DIM_P + col0 + 16 * t + n] = (__bf16)acc[t][r];
}

// ---------------------------------------------------------------------------
// Fused attention: scores = Q@K^T /32 -> softmax -> @V -> +jf
// One block (8 waves, 256 thr) per 16 rows of B. Q block async-staged to LDS.
// ---------------------------------------------------------------------------
#define QS_LD 776   /* 768 + 8 pad elems: 1552B row -> bank offset 4/row */

__global__ void ccim_attn(const float* __restrict__ jf,
                          const __bf16* __restrict__ Qb,
                          const __bf16* __restrict__ Kb,
                          const __bf16* __restrict__ Vt,
                          float* __restrict__ out) {
  __shared__ __bf16 q_s[16 * QS_LD];   // 24.25 KB staged Q block
  __shared__ float  sc[16][260];       // raw scores (padded)
  __shared__ __bf16 attn_s[16][264];   // un-normalized exp() weights, bf16
  __shared__ float  red[16][16];
  __shared__ float  rowmax[16];
  __shared__ float  rowinv[16];

  const int t    = threadIdx.x;
  const int w    = t >> 5;             // wave 0..7
  const int lane = t & 31;
  const int n    = lane & 15;
  const int kh   = lane >> 4;
  const int row0 = blockIdx.x * 16;

  // ---- async-stage Q[row0:+16][0:768] into LDS (16B chunks, padded rows) --
  {
    const unsigned lds_base = (unsigned)(size_t)(&q_s[0]);
#pragma unroll
    for (int it = 0; it < 6; ++it) {             // 1536 chunks / 256 threads
      int c  = t + it * 256;
      int r  = c / 96, cc = c - r * 96;          // 96 x 16B chunks per row
      unsigned lds = lds_base + (unsigned)(r * (QS_LD * 2) + cc * 16);
      unsigned long long ga =
          (unsigned long long)(size_t)(Qb + (row0 + r) * DIM_P + cc * 8);
      asm volatile("global_load_async_to_lds_b128 %0, %1, off"
                   :: "v"(lds), "v"(ga) : "memory");
    }
    asm volatile("s_wait_asynccnt 0x0" ::: "memory");
  }
  __syncthreads();

  // ---- scores: wave w owns K-columns [32w, 32w+32); reduce over P=768 ----
  v8f s0 = {}, s1 = {};
  for (int k0 = 0; k0 < DIM_P; k0 += 64) {         // 12 iters, no remainder
    v16bf a0  = loadA_bf16(q_s, QS_LD, n, k0, kh);
    v16bf c0  = loadB_rows_bf16(Kb, DIM_P, 32 * w + n, k0, kh);
    v16bf d0  = loadB_rows_bf16(Kb, DIM_P, 32 * w + 16 + n, k0, kh);
    v16bf a1  = loadA_bf16(q_s, QS_LD, n, k0 + 32, kh);
    v16bf c1  = loadB_rows_bf16(Kb, DIM_P, 32 * w + n, k0 + 32, kh);
    v16bf d1  = loadB_rows_bf16(Kb, DIM_P, 32 * w + 16 + n, k0 + 32, kh);
    s0 = wmma_bf16(a0, c0, s0);
    s1 = wmma_bf16(a0, d0, s1);
    s0 = wmma_bf16(a1, c1, s0);
    s1 = wmma_bf16(a1, d1, s1);
  }
  {
    const int mo = kh * 8;
#pragma unroll
    for (int r = 0; r < 8; ++r) {
      sc[mo + r][32 * w + n]      = s0[r];
      sc[mo + r][32 * w + 16 + n] = s1[r];
    }
  }
  __syncthreads();

  // ---- softmax over 256 confounders: 16 threads per row ----
  const int srow = t >> 4, seg = t & 15;
  float lmax = -3.4e38f;
#pragma unroll
  for (int i = 0; i < 16; ++i) lmax = fmaxf(lmax, sc[srow][seg * 16 + i]);
  red[srow][seg] = lmax;
  __syncthreads();
  if (seg == 0) {
    float mx = red[srow][0];
#pragma unroll
    for (int i = 1; i < 16; ++i) mx = fmaxf(mx, red[srow][i]);
    rowmax[srow] = mx;
  }
  __syncthreads();
  const float mx = rowmax[srow];
  float lsum = 0.f;
#pragma unroll
  for (int i = 0; i < 16; ++i) {
    // scale 1/sqrt(1024)=1/32 folded into the exponent
    float e = __expf((sc[srow][seg * 16 + i] - mx) * 0.03125f);
    lsum += e;
    attn_s[srow][seg * 16 + i] = (__bf16)e;
  }
  red[srow][seg] = lsum;
  __syncthreads();
  if (seg == 0) {
    float s = 0.f;
#pragma unroll
    for (int i = 0; i < 16; ++i) s += red[srow][i];
    rowinv[srow] = 1.0f / s;           // normalization folded into epilogue
  }
  __syncthreads();

  // ---- g_z = attn @ V : wave w owns cols {16w + 128t}, reduce K=256 ----
  // Per k-step: batch A + all 8 B loads first, then 8 WMMAs -> descending
  // s_wait_loadcnt thresholds overlap early WMMAs with in-flight B tiles.
  v8f acc[8] = {};
  for (int k0 = 0; k0 < DIM_K; k0 += 32) {
    v16bf a = loadA_bf16(&attn_s[0][0], 264, n, k0, kh);
    v16bf b[8];
#pragma unroll
    for (int tt = 0; tt < 8; ++tt)
      b[tt] = loadB_rows_bf16(Vt, DIM_K, 16 * w + 128 * tt + n, k0, kh);
#pragma unroll
    for (int tt = 0; tt < 8; ++tt) acc[tt] = wmma_bf16(a, b[tt], acc[tt]);
  }

  const int mo = kh * 8;
#pragma unroll
  for (int tt = 0; tt < 8; ++tt) {
    const int d = 16 * w + 128 * tt + n;
#pragma unroll
    for (int r = 0; r < 8; ++r) {
      const int m  = mo + r;
      const int gr = row0 + m;
      out[gr * DIM_D + d] = jf[gr * DIM_D + d] + acc[tt][r] * rowinv[m];
    }
  }
}

// ---------------------------------------------------------------------------
extern "C" void kernel_launch(void* const* d_in, const int* in_sizes, int n_in,
                              void* d_out, int out_size, void* d_ws,
                              size_t ws_size, hipStream_t stream) {
  (void)in_sizes; (void)n_in; (void)out_size; (void)ws_size;
  const float* jf    = (const float*)d_in[0];   // [1024,1024]
  const float* conf  = (const float*)d_in[1];   // [256,1024]
  const float* prior = (const float*)d_in[2];   // [256,1]
  const float* Wq    = (const float*)d_in[3];   // [1024,768]
  const float* Wk    = (const float*)d_in[4];   // [1024,768]
  float* out = (float*)d_out;                   // [1024,1024]

  // bf16 workspace (L2-resident), element offsets:
  //   Q[1024x768] | K[256x768] | Vt[1024x256] | Wqt | Wkt | Jb | Cb
  __bf16* Qb  = (__bf16*)d_ws;
  __bf16* Kb  = Qb  + DIM_B * DIM_P;
  __bf16* Vt  = Kb  + DIM_K * DIM_P;
  __bf16* Wqt = Vt  + DIM_D * DIM_K;
  __bf16* Wkt = Wqt + DIM_P * DIM_D;
  __bf16* Jb  = Wkt + DIM_P * DIM_D;
  __bf16* Cb  = Jb  + DIM_B * DIM_D;

  ccim_cvt  <<<(DIM_B * DIM_D) / 256, 256, 0, stream>>>(jf, Jb);
  ccim_wprep<<<(DIM_D * DIM_P) / 256, 256, 0, stream>>>(Wq, Wqt);
  ccim_wprep<<<(DIM_D * DIM_P) / 256, 256, 0, stream>>>(Wk, Wkt);
  ccim_vprep<<<(DIM_K * DIM_D) / 256, 256, 0, stream>>>(conf, prior, Vt, Cb);
  ccim_proj<<<96, 256, 0, stream>>>(Jb, Wqt, Qb, DIM_B);   // 64*12 waves
  ccim_proj<<<24, 256, 0, stream>>>(Cb, Wkt, Kb, DIM_K);   // 16*12 waves
  ccim_attn<<<DIM_B / 16, 256, 0, stream>>>(jf, Qb, Kb, Vt, out);
}
